// MSDeform_OnScan_v1_62079457296617
// MI455X (gfx1250) — compile-verified, matrix operand order checked by reading, not run on hardware
//
#include <hip/hip_runtime.h>
#include <hip/hip_bf16.h>
#include <math.h>

#define NQ     5440
#define DM     256
#define NH     8
#define NL     4
#define NP     4
#define DS     16
#define HD     32
#define SEQ    160   // NH*NL*(NP+1)

typedef _Float16 v16h __attribute__((ext_vector_type(16)));
typedef float    v8f  __attribute__((ext_vector_type(8)));

// ---------------------------------------------------------------------------
// Kernel 1: depthwise 3x3 conv per pyramid level.
// input_flatten is (hw, 256) pixel-major; output stored (pix, 256) so that the
// sampler reads one head's 32 channels as a single coalesced 128B burst.
// ---------------------------------------------------------------------------
__global__ void __launch_bounds__(256) conv_dw3x3(
    const float* __restrict__ in, const float* __restrict__ w,
    const float* __restrict__ b, float* __restrict__ out)
{
    int pix = blockIdx.x;
    int c   = threadIdx.x;
    int H, W, ls;
    if      (pix < 4096) { H = 64; W = 64; ls = 0;    }
    else if (pix < 5120) { H = 32; W = 32; ls = 4096; }
    else if (pix < 5376) { H = 16; W = 16; ls = 5120; }
    else                 { H = 8;  W = 8;  ls = 5376; }
    int p = pix - ls;
    int y = p / W;
    int x = p - y * W;
    float acc = b[c];
#pragma unroll
    for (int dy = -1; dy <= 1; ++dy) {
#pragma unroll
        for (int dx = -1; dx <= 1; ++dx) {
            int yy = y + dy, xx = x + dx;
            if (yy >= 0 && yy < H && xx >= 0 && xx < W) {
                acc += w[c * 9 + (dy + 1) * 3 + (dx + 1)] *
                       in[(size_t)(ls + yy * W + xx) * DM + c];
            }
        }
    }
    out[(size_t)pix * DM + c] = acc;
}

// ---------------------------------------------------------------------------
// Kernel 2/4: C = A(MxK) * W(NxK)^T + bias, K=N=256, f16 WMMA, f32 accumulate.
// 8 waves/WG in 2x4 layout -> 32(M) x 64(N) per WG.
// A 32x32 chunk and W 64x32 chunk are staged in LDS via CDNA5
// global_load_async_to_lds_b128 (ASYNCcnt), then fragments are gathered from
// LDS following the ISA 7.12.2 16-bit layouts:
//   lane<16 : K = {kb+0..7, kb+16..23}, lane>=16 : K = {kb+8..15, kb+24..31}
// K-loop fully unrolled -> 8 static v_wmma issues per wave.
// ---------------------------------------------------------------------------
__global__ void __launch_bounds__(256) gemm256_wmma(
    const float* __restrict__ A, const float* __restrict__ W,
    const float* __restrict__ bias, float* __restrict__ C)
{
    __shared__ float As[32][33];   // A chunk: 32 rows x 32 K (padded)
    __shared__ float Ws[64][33];   // W chunk: 64 rows x 32 K (padded)

    const int t    = threadIdx.x;
    const int lane = t & 31;
    const int wv   = t >> 5;

    const int mTile = blockIdx.x * 32;
    const int nTile = blockIdx.y * 64;

    // uniform SGPR-pair bases for the async-to-LDS SADDR form
    const unsigned long long aBase = (unsigned long long)(A + (size_t)mTile * DM);
    const unsigned long long wBase = (unsigned long long)(W + (size_t)nTile * DM);

    // per-thread staging assignment (b128 = 4 floats)
    const int arow = t >> 3;           // 0..31
    const int aseg = (t & 7) * 4;      // 0,4,..,28
    const unsigned aLds = (unsigned)(unsigned long long)&As[arow][aseg];
    const int wrow0 = t >> 3;          // 0..31
    const int wrow1 = (t + 256) >> 3;  // 32..63
    const unsigned wLds0 = (unsigned)(unsigned long long)&Ws[wrow0][aseg];
    const unsigned wLds1 = (unsigned)(unsigned long long)&Ws[wrow1][aseg];

    const int r  = lane & 15;
    const int c0 = (lane >> 4) * 8;
    const int ma = (wv >> 2) * 16 + r;   // row in As
    const int nb = (wv & 3) * 16 + r;    // row in Ws

    v8f acc = {};
#pragma unroll
    for (int kb = 0; kb < DM; kb += 32) {
        // ---- async stage: global -> LDS, tracked by ASYNCcnt
        unsigned aOff  = (unsigned)((arow  * DM + kb) * 4) + (unsigned)(aseg * 4);
        unsigned wOff0 = (unsigned)((wrow0 * DM + kb) * 4) + (unsigned)(aseg * 4);
        unsigned wOff1 = (unsigned)((wrow1 * DM + kb) * 4) + (unsigned)(aseg * 4);
        asm volatile("global_load_async_to_lds_b128 %0, %1, %2"
                     :: "v"(aLds), "v"(aOff), "s"(aBase) : "memory");
        asm volatile("global_load_async_to_lds_b128 %0, %1, %2"
                     :: "v"(wLds0), "v"(wOff0), "s"(wBase) : "memory");
        asm volatile("global_load_async_to_lds_b128 %0, %1, %2"
                     :: "v"(wLds1), "v"(wOff1), "s"(wBase) : "memory");
        asm volatile("s_wait_asynccnt 0x0" ::: "memory");
        __syncthreads();

        // ---- gather WMMA fragments from LDS, cvt f32->f16
        v16h af, bf;
#pragma unroll
        for (int j = 0; j < 8; ++j) {
            af[j]     = (_Float16)As[ma][c0 + j];
            af[j + 8] = (_Float16)As[ma][16 + c0 + j];
            bf[j]     = (_Float16)Ws[nb][c0 + j];
            bf[j + 8] = (_Float16)Ws[nb][16 + c0 + j];
        }
        acc = __builtin_amdgcn_wmma_f32_16x16x32_f16(
            false, af, false, bf, (short)0, acc, false, false);
        __syncthreads();   // protect LDS before next chunk overwrites
    }

    const int hi = (lane >> 4) * 8;
#pragma unroll
    for (int j = 0; j < 8; ++j) {
        int m = mTile + (wv >> 2) * 16 + hi + j;
        int n = nTile + (wv & 3) * 16 + r;
        C[(size_t)m * DM + n] = acc[j] + bias[n];
    }
}

// ---------------------------------------------------------------------------
// Kernel 3: fused bilinear sampling + x_dbl + S6 scan, one query per block.
// Avoids materializing the 111 MB ss tensor: everything lives in ~50KB LDS.
// ---------------------------------------------------------------------------
__global__ void __launch_bounds__(256) msdeform_scan(
    const float* __restrict__ off,   const float* __restrict__ refp,
    const float* __restrict__ val,   const float* __restrict__ Wx,
    const float* __restrict__ Wdt,   const float* __restrict__ b_dt,
    const float* __restrict__ A_log, const float* __restrict__ Dp,
    float* __restrict__ Y)
{
    __shared__ float xbuf[SEQ][33];     // x[d][l] per query  (padded)
    __shared__ float xdbuf[SEQ][36];    // x_dbl[r][l], r in 0..33 (padded)
    __shared__ float wxs[34 * 32];
    __shared__ float ybuf[DM];

    const int q    = blockIdx.x;
    const int t    = threadIdx.x;
    const int lane = t & 31;
    const int wv   = t >> 5;

    for (int i = t; i < 34 * 32; i += 256) wxs[i] = Wx[i];

    // ---- Phase A: bilinear sampling. wave wv handles steps l = wv + 8*i.
    const int HWs[4] = {64, 32, 16, 8};
    const int Lst[4] = {0, 4096, 5120, 5376};
#pragma unroll 4
    for (int i = 0; i < 20; ++i) {
        int l   = wv + 8 * i;
        int hd  = l / 20;
        int rem = l - hd * 20;
        int lvl = rem / 5;
        int k   = rem - lvl * 5;
        int Wl  = HWs[lvl], Hl = HWs[lvl], ls = Lst[lvl];

        float rx = refp[(q * NL + lvl) * 2 + 0];
        float ry = refp[(q * NL + lvl) * 2 + 1];
        float lx, ly;
        if (k < NP) {
            int oi = q * DM + (((hd * NL + lvl) * NP + k) * 2);
            lx = rx + off[oi]     / (float)Wl;
            ly = ry + off[oi + 1] / (float)Hl;
        } else { lx = rx; ly = ry; }

        float gx = lx * (float)Wl - 0.5f;
        float gy = ly * (float)Hl - 0.5f;
        float x0f = floorf(gx), y0f = floorf(gy);
        int   x0  = (int)x0f,   y0  = (int)y0f;
        float wx1 = gx - x0f,   wy1 = gy - y0f;
        float wx0 = 1.f - wx1,  wy0 = 1.f - wy1;

        int ch = hd * HD + lane;                 // coalesced across the wave
        float v00 = 0.f, v10 = 0.f, v01 = 0.f, v11 = 0.f;
        if (y0 >= 0 && y0 < Hl) {
            if (x0 >= 0 && x0 < Wl)
                v00 = val[(size_t)(ls + y0 * Wl + x0) * DM + ch];
            if (x0 + 1 >= 0 && x0 + 1 < Wl)
                v10 = val[(size_t)(ls + y0 * Wl + x0 + 1) * DM + ch];
        }
        if (y0 + 1 >= 0 && y0 + 1 < Hl) {
            if (x0 >= 0 && x0 < Wl)
                v01 = val[(size_t)(ls + (y0 + 1) * Wl + x0) * DM + ch];
            if (x0 + 1 >= 0 && x0 + 1 < Wl)
                v11 = val[(size_t)(ls + (y0 + 1) * Wl + x0 + 1) * DM + ch];
        }
        xbuf[l][lane] = v00 * wx0 * wy0 + v10 * wx1 * wy0 +
                        v01 * wx0 * wy1 + v11 * wx1 * wy1;
    }
    __syncthreads();

    // ---- Phase B: x_dbl[r][l] = sum_d Wx[r][d] * x[d][l]   (5440 dots of 32)
    for (int idx = t; idx < SEQ * 34; idx += 256) {
        int l = idx / 34;
        int r = idx - l * 34;
        float acc = 0.f;
#pragma unroll
        for (int d = 0; d < HD; ++d) acc += wxs[r * HD + d] * xbuf[l][d];
        xdbuf[l][r] = acc;
    }
    __syncthreads();

    // ---- Phase C: sequential S6 scan. Each wave owns 2 channels x 16 states
    // per pass (lane = d_local*16 + n); two passes cover all 32 channels.
    for (int pass = 0; pass < 2; ++pass) {
        int d = pass * 16 + wv * 2 + (lane >> 4);
        int n = lane & 15;
        float Adn = -expf(A_log[d * DS + n]);
        float w0  = Wdt[d * 2 + 0];
        float w1  = Wdt[d * 2 + 1];
        float bd  = b_dt[d];
        float Dd  = Dp[d];
        float hs  = 0.f;
        float accHead = 0.f;
        for (int l = 0; l < SEQ; ++l) {
            float z  = w0 * xdbuf[l][0] + w1 * xdbuf[l][1] + bd;
            float dl = (z > 20.f) ? z : log1pf(expf(z));      // softplus
            float xv = xbuf[l][d];
            float bl = xdbuf[l][2 + n];
            hs = expf(dl * Adn) * hs + (dl * xv) * bl;
            if ((l % 5) == 4) {                               // k == NP sample
                float cl   = xdbuf[l][2 + DS + n];
                float part = hs * cl;
                part += __shfl_xor(part, 1);
                part += __shfl_xor(part, 2);
                part += __shfl_xor(part, 4);
                part += __shfl_xor(part, 8);                  // sum over n
                accHead += part + Dd * xv;                    // + D*x term
                if ((l % 20) == 19) {                         // level sum done
                    if (n == 0) ybuf[d * NH + (l / 20)] = accHead;
                    accHead = 0.f;
                }
            }
        }
    }
    __syncthreads();

    // column order of reference reshape is d*8 + head — ybuf matches.
    Y[(size_t)q * DM + t] = ybuf[t];
}

// ---------------------------------------------------------------------------
extern "C" void kernel_launch(void* const* d_in, const int* in_sizes, int n_in,
                              void* d_out, int out_size, void* d_ws, size_t ws_size,
                              hipStream_t stream) {
    (void)in_sizes; (void)n_in; (void)out_size; (void)ws_size;

    const float* query  = (const float*)d_in[0];
    const float* refp   = (const float*)d_in[1];
    const float* inflat = (const float*)d_in[2];
    // d_in[3] (spatial shapes) / d_in[4] (level starts) are compile-time consts
    const float* W_off  = (const float*)d_in[5];
    const float* b_off  = (const float*)d_in[6];
    const float* conv_w = (const float*)d_in[7];
    const float* conv_b = (const float*)d_in[8];
    const float* Wx     = (const float*)d_in[9];
    const float* Wdt    = (const float*)d_in[10];
    const float* b_dt   = (const float*)d_in[11];
    const float* A_log  = (const float*)d_in[12];
    const float* Dvec   = (const float*)d_in[13];
    const float* Wo     = (const float*)d_in[14];
    const float* bo     = (const float*)d_in[15];

    char* ws = (char*)d_ws;
    const size_t SZ = (size_t)NQ * DM * sizeof(float);   // 5.57 MB each
    float* off  = (float*)(ws);
    float* valb = (float*)(ws + SZ);
    float* Y    = (float*)(ws + 2 * SZ);

    // 1) depthwise conv into (pix, 256) layout
    conv_dw3x3<<<NQ, 256, 0, stream>>>(inflat, conv_w, conv_b, valb);
    // 2) offsets GEMM via WMMA: off = query @ W_off^T + b_off
    gemm256_wmma<<<dim3(NQ / 32, DM / 64), 256, 0, stream>>>(query, W_off, b_off, off);
    // 3) fused sampling + x_dbl + S6 scan (one query per block)
    msdeform_scan<<<NQ, 256, 0, stream>>>(off, refp, valb, Wx, Wdt, b_dt,
                                          A_log, Dvec, Y);
    // 4) output projection via WMMA: out = Y @ Wo^T + bo
    gemm256_wmma<<<dim3(NQ / 32, DM / 64), 256, 0, stream>>>(Y, Wo, bo, (float*)d_out);
}
